// LinearAttention_23210003267591
// MI455X (gfx1250) — matrix operand order repaired and checked by reading, not compiled, not run on hardware
//
#include <hip/hip_runtime.h>
#include <hip/hip_bf16.h>
#include <stdint.h>

// ---------------------------------------------------------------------------
// Types for CDNA5 WMMA (wave32, v_wmma_f32_16x16x32_bf16)
// ---------------------------------------------------------------------------
typedef unsigned int  u32;
typedef unsigned short u16;
typedef u32  __attribute__((ext_vector_type(4)))  u32x4;
typedef __bf16 bf16_t;
typedef bf16_t __attribute__((ext_vector_type(16))) v16bf;
typedef float  __attribute__((ext_vector_type(8)))  v8f;

union FragAB { v16bf v; u32x4 u[2]; };   // 32 bytes: one WMMA A or B operand
union Pack8  { u32x4 u; u16 s[8]; };

__device__ __forceinline__ u16 f2bf(float f) {
  u32 x = __float_as_uint(f);
  u32 r = 0x7FFFu + ((x >> 16) & 1u);      // round-to-nearest-even
  return (u16)((x + r) >> 16);
}

__device__ __forceinline__ v8f wmma_bf16(const v16bf a, const v16bf b, v8f c) {
  // D = A(16x32 bf16) * B(32x16 bf16) + C(16x16 f32)
  return __builtin_amdgcn_wmma_f32_16x16x32_bf16(
      /*neg_a=*/false, a, /*neg_b=*/false, b,
      /*c_mod=*/(short)0, c, /*reuse_a=*/false, /*reuse_b=*/false);
}

// Async global->LDS copy (CDNA5 GLOBAL_LOAD_ASYNC_TO_LDS_B128, ASYNCcnt).
// lds_off: workgroup-relative LDS byte address (low 32 bits of flat shared ptr).
__device__ __forceinline__ void async_g2l_b128(u32 lds_off, const void* gaddr) {
  asm volatile("global_load_async_to_lds_b128 %0, %1, off"
               :: "v"(lds_off), "v"(gaddr) : "memory");
}
// Async loads retire in order: waiting ASYNCcnt<=N with N ops of the *next*
// tile in flight guarantees the *previous* tile's transfers have landed.
__device__ __forceinline__ void wait_async_le4() {
  asm volatile("s_wait_asynccnt 0x4" ::: "memory");
}
__device__ __forceinline__ void wait_async_le1() {
  asm volatile("s_wait_asynccnt 0x1" ::: "memory");
}
__device__ __forceinline__ void wait_async_le0() {
  asm volatile("s_wait_asynccnt 0x0" ::: "memory");
}
__device__ __forceinline__ u32 lds_addr(const void* p) {
  return (u32)(uintptr_t)p;   // flat LDS address: low 32 bits are WG-relative
}

// Problem constants (match reference)
#define BATCH 2
#define SEQ   2048
#define EMB   1024
#define NHEAD 16
#define HD    64
#define MROWS (BATCH * SEQ)   // 4096

// ---------------------------------------------------------------------------
// f32 -> bf16 conversion
// ---------------------------------------------------------------------------
__global__ void k_cvt_bf16(const float* __restrict__ src, u16* __restrict__ dst, int n) {
  int i = blockIdx.x * blockDim.x + threadIdx.x;
  if (i < n) dst[i] = f2bf(src[i]);
}

// ---------------------------------------------------------------------------
// GEMM: C[M,N] = A[M,K](bf16) * W[N,K]^T(bf16) + bias[N](f32)
// 256 thr (8 waves), tile 128x128, K slab 32, double-buffered async DMA to
// LDS overlapping WMMA. Wave (wm 0..1, wn 0..3) computes 64x32 = 4x2 frags.
// ---------------------------------------------------------------------------
template <bool OUTF32>
__global__ __launch_bounds__(256) void k_gemm(const u16* __restrict__ A,
                                              const u16* __restrict__ W,
                                              const float* __restrict__ bias,
                                              void* __restrict__ Cout,
                                              int M, int N, int K) {
  __shared__ __align__(16) u16 sA[2][128 * 32];
  __shared__ __align__(16) u16 sB[2][128 * 32];

  const int tid = threadIdx.x;
  const int w   = tid >> 5;
  const int lane = tid & 31;
  const int lr  = lane & 15;     // lane within half
  const int hl  = lane >> 4;     // which half of the wave
  const int wm  = w & 1, wn = w >> 1;
  const int bm  = blockIdx.y * 128;
  const int bn  = blockIdx.x * 128;

  // Per-thread DMA chunk coordinates (4 async b128 per thread per K-slab)
  auto issue_tile = [&](int k0, int buf) {
#pragma unroll
    for (int i = 0; i < 2; ++i) {
      int c   = tid + i * 256;
      int row = c >> 2;
      int col = (c & 3) * 8;
      async_g2l_b128(lds_addr(&sA[buf][row * 32 + col]),
                     &A[(size_t)(bm + row) * K + k0 + col]);
      async_g2l_b128(lds_addr(&sB[buf][row * 32 + col]),
                     &W[(size_t)(bn + row) * K + k0 + col]);
    }
  };

  v8f acc[4][2];
#pragma unroll
  for (int mi = 0; mi < 4; ++mi)
#pragma unroll
    for (int ni = 0; ni < 2; ++ni) acc[mi][ni] = {};

  const int NT = K / 32;
  issue_tile(0, 0);                 // prologue: DMA tile 0 into buffer 0

  for (int kt = 0; kt < NT; ++kt) {
    const int buf = kt & 1;
    __syncthreads();                // readers of buf^1 (tile kt-1) are done
    if (kt + 1 < NT) {
      issue_tile((kt + 1) * 32, buf ^ 1);
      wait_async_le4();             // tile kt landed; tile kt+1 still in flight
    } else {
      wait_async_le0();
    }
    __syncthreads();                // all waves' tile-kt data visible

    FragAB af[4], bfr[2];
#pragma unroll
    for (int mi = 0; mi < 4; ++mi) {
      // A layout 16x32: lanes 0-15 row M=lr K{0..7,16..23}; lanes 16-31 K{8..15,24..31}
      int row = wm * 64 + mi * 16 + lr;
      af[mi].u[0] = *(const u32x4*)&sA[buf][row * 32 + 8 * hl];
      af[mi].u[1] = *(const u32x4*)&sA[buf][row * 32 + 16 + 8 * hl];
    }
#pragma unroll
    for (int ni = 0; ni < 2; ++ni) {
      // B layout 32x16: lane col N=lr, K{0..7,16..23} for lanes 0-15; K{8..15,24..31} for lanes 16-31
      int col = wn * 32 + ni * 16 + lr;
      bfr[ni].u[0] = *(const u32x4*)&sB[buf][col * 32 + 8 * hl];
      bfr[ni].u[1] = *(const u32x4*)&sB[buf][col * 32 + 16 + 8 * hl];
    }
#pragma unroll
    for (int mi = 0; mi < 4; ++mi)
#pragma unroll
      for (int ni = 0; ni < 2; ++ni)
        acc[mi][ni] = wmma_bf16(af[mi].v, bfr[ni].v, acc[mi][ni]);
  }

  // Epilogue: C row = r + 8*hl (per 32-bit C/D layout), col = lr
#pragma unroll
  for (int mi = 0; mi < 4; ++mi) {
#pragma unroll
    for (int ni = 0; ni < 2; ++ni) {
      int n = bn + wn * 32 + ni * 16 + lr;
      float bv = bias[n];
#pragma unroll
      for (int r = 0; r < 8; ++r) {
        int m = bm + wm * 64 + mi * 16 + r + 8 * hl;
        float v = acc[mi][ni][r] + bv;
        if constexpr (OUTF32)
          ((float*)Cout)[(size_t)m * N + n] = v;
        else
          ((u16*)Cout)[(size_t)m * N + n] = f2bf(v);
      }
    }
  }
}

// ---------------------------------------------------------------------------
// Flash attention: grid (S/128, NHEAD, BATCH), block 256 (8 waves).
// Wave w handles 16 query rows m0 = mb*128 + w*16 over all keys <= causal bound.
// K tile double-buffered via async DMA; V transposed through VGPRs into LDS.
// ---------------------------------------------------------------------------
__global__ __launch_bounds__(256) void k_flash(const u16* __restrict__ Qp,
                                               const u16* __restrict__ Kp,
                                               const u16* __restrict__ Vp,
                                               const unsigned char* __restrict__ pm,
                                               u16* __restrict__ AO) {
  __shared__ __align__(16) u16 sK[2][32 * 64];   // [buf][key][dim]
  __shared__ __align__(16) u16 sVt[64 * 32];     // [dim][key]  (transposed V)
  __shared__ __align__(16) u16 sP[8 * 16 * 32];  // per-wave P tiles (16x32)

  const int tid = threadIdx.x;
  const int w = tid >> 5, lane = tid & 31, lr = lane & 15, hl = lane >> 4;
  const int mb = blockIdx.x, h = blockIdx.y, b = blockIdx.z;
  const int m0 = mb * 128 + w * 16;
  const float SCALE = 0.03125f;   // EMB^-0.5 = 1/32

  const int kk = tid >> 3;        // DMA chunk coords (1 async b128 per thread)
  const int cc = (tid & 7) * 8;
  auto issue_k = [&](int j, int buf) {
    async_g2l_b128(lds_addr(&sK[buf][kk * 64 + cc]),
                   Kp + (size_t)(b * SEQ + j + kk) * EMB + h * HD + cc);
  };

  // Q A-fragments: rows m0+lr (both halves), head dims 0..31 and 32..63
  FragAB aq0, aq1;
  {
    const u16* qr = Qp + (size_t)(b * SEQ + m0 + lr) * EMB + h * HD;
    aq0.u[0] = *(const u32x4*)(qr + 8 * hl);
    aq0.u[1] = *(const u32x4*)(qr + 16 + 8 * hl);
    aq1.u[0] = *(const u32x4*)(qr + 32 + 8 * hl);
    aq1.u[1] = *(const u32x4*)(qr + 48 + 8 * hl);
  }

  v8f o0 = {}, o1 = {}, o2 = {}, o3 = {};
  float mrun[8], lrun[8];
#pragma unroll
  for (int r = 0; r < 8; ++r) { mrun[r] = -1e30f; lrun[r] = 0.0f; }

  const int jend = (mb + 1) * 128;   // causal bound, uniform for the workgroup
  issue_k(0, 0);                     // prologue: DMA key block 0

  for (int j = 0; j < jend; j += 32) {
    const int buf = (j >> 5) & 1;
    __syncthreads();                 // prior compute's LDS reads are done
    {
      // V block: load + transpose into sVt through VGPRs
      size_t rowoff = (size_t)(b * SEQ + j + kk) * EMB + h * HD + cc;
      Pack8 pv; pv.u = *(const u32x4*)(Vp + rowoff);
#pragma unroll
      for (int i = 0; i < 8; ++i) sVt[(cc + i) * 32 + kk] = pv.s[i];
      if (j + 32 < jend)
        __builtin_prefetch(Vp + rowoff + (size_t)32 * EMB, 0, 1);
    }
    if (j + 32 < jend) {
      issue_k(j + 32, buf ^ 1);      // overlap next K block with this compute
      wait_async_le1();              // block j landed; block j+32 in flight
    } else {
      wait_async_le0();
    }
    __syncthreads();                 // sVt stores + all K chunks visible

    // S = Q * K^T for two 16-key groups (keys j+lr and j+16+lr per lane)
    v8f s0 = {}, s1 = {};
    {
      const u16* kb = &sK[buf][0];
      FragAB bk;
      bk.u[0] = *(const u32x4*)&kb[lr * 64 + 8 * hl];
      bk.u[1] = *(const u32x4*)&kb[lr * 64 + 16 + 8 * hl];
      s0 = wmma_bf16(aq0.v, bk.v, s0);
      bk.u[0] = *(const u32x4*)&kb[lr * 64 + 32 + 8 * hl];
      bk.u[1] = *(const u32x4*)&kb[lr * 64 + 48 + 8 * hl];
      s0 = wmma_bf16(aq1.v, bk.v, s0);
      bk.u[0] = *(const u32x4*)&kb[(16 + lr) * 64 + 8 * hl];
      bk.u[1] = *(const u32x4*)&kb[(16 + lr) * 64 + 16 + 8 * hl];
      s1 = wmma_bf16(aq0.v, bk.v, s1);
      bk.u[0] = *(const u32x4*)&kb[(16 + lr) * 64 + 32 + 8 * hl];
      bk.u[1] = *(const u32x4*)&kb[(16 + lr) * 64 + 48 + 8 * hl];
      s1 = wmma_bf16(aq1.v, bk.v, s1);
    }

    const int key0 = j + lr, key1 = j + 16 + lr;
    const bool pad0 = pm[(size_t)b * SEQ + key0] != 0;
    const bool pad1 = pm[(size_t)b * SEQ + key1] != 0;

    // Online softmax per row (row = r + 8*hl; 16 cols live across lanes of a half)
#pragma unroll
    for (int r = 0; r < 8; ++r) {
      int qrw = m0 + r + 8 * hl;
      bool k0ok = (key0 <= qrw) && pad0;
      bool k1ok = (key1 <= qrw) && pad1;
      float e0 = k0ok ? s0[r] * SCALE : -1e30f;
      float e1 = k1ok ? s1[r] * SCALE : -1e30f;
      float bmx = fmaxf(e0, e1);
      bmx = fmaxf(bmx, __shfl_xor(bmx, 1, 32));
      bmx = fmaxf(bmx, __shfl_xor(bmx, 2, 32));
      bmx = fmaxf(bmx, __shfl_xor(bmx, 4, 32));
      bmx = fmaxf(bmx, __shfl_xor(bmx, 8, 32));
      float nm = fmaxf(mrun[r], bmx);
      float al = __expf(mrun[r] - nm);
      float p0 = k0ok ? __expf(e0 - nm) : 0.0f;
      float p1 = k1ok ? __expf(e1 - nm) : 0.0f;
      float ps = p0 + p1;
      ps += __shfl_xor(ps, 1, 32);
      ps += __shfl_xor(ps, 2, 32);
      ps += __shfl_xor(ps, 4, 32);
      ps += __shfl_xor(ps, 8, 32);
      lrun[r] = lrun[r] * al + ps;
      mrun[r] = nm;
      o0[r] *= al; o1[r] *= al; o2[r] *= al; o3[r] *= al;
      int prow = r + 8 * hl;
      sP[w * 512 + prow * 32 + lr]      = f2bf(p0);
      sP[w * 512 + prow * 32 + 16 + lr] = f2bf(p1);
    }

    // Same-wave LDS RAW: force ordering + hardware wait before re-reading P
    asm volatile("s_wait_dscnt 0" ::: "memory");

    // O += P(16x32) * V(32x64): one A frag, four B frags from transposed V
    FragAB ap, bv;
    ap.u[0] = *(const u32x4*)&sP[w * 512 + lr * 32 + 8 * hl];
    ap.u[1] = *(const u32x4*)&sP[w * 512 + lr * 32 + 16 + 8 * hl];
    bv.u[0] = *(const u32x4*)&sVt[(0 + lr) * 32 + 8 * hl];
    bv.u[1] = *(const u32x4*)&sVt[(0 + lr) * 32 + 16 + 8 * hl];
    o0 = wmma_bf16(ap.v, bv.v, o0);
    bv.u[0] = *(const u32x4*)&sVt[(16 + lr) * 32 + 8 * hl];
    bv.u[1] = *(const u32x4*)&sVt[(16 + lr) * 32 + 16 + 8 * hl];
    o1 = wmma_bf16(ap.v, bv.v, o1);
    bv.u[0] = *(const u32x4*)&sVt[(32 + lr) * 32 + 8 * hl];
    bv.u[1] = *(const u32x4*)&sVt[(32 + lr) * 32 + 16 + 8 * hl];
    o2 = wmma_bf16(ap.v, bv.v, o2);
    bv.u[0] = *(const u32x4*)&sVt[(48 + lr) * 32 + 8 * hl];
    bv.u[1] = *(const u32x4*)&sVt[(48 + lr) * 32 + 16 + 8 * hl];
    o3 = wmma_bf16(ap.v, bv.v, o3);
  }

  // Normalize and store (merged-head layout [B,S,EMB], bf16)
#pragma unroll
  for (int r = 0; r < 8; ++r) {
    float inv = 1.0f / lrun[r];
    int qrw = m0 + r + 8 * hl;
    size_t base = (size_t)(b * SEQ + qrw) * EMB + h * HD + lr;
    AO[base + 0]  = f2bf(o0[r] * inv);
    AO[base + 16] = f2bf(o1[r] * inv);
    AO[base + 32] = f2bf(o2[r] * inv);
    AO[base + 48] = f2bf(o3[r] * inv);
  }
}

// ---------------------------------------------------------------------------
// Launcher
// Inputs: 0:q 1:k 2:v 3:tri_mask 4:padding_mask 5:Wq 6:bq 7:Wk 8:bk 9:Wv 10:bv 11:Wo 12:bo
// ---------------------------------------------------------------------------
extern "C" void kernel_launch(void* const* d_in, const int* in_sizes, int n_in,
                              void* d_out, int out_size, void* d_ws, size_t ws_size,
                              hipStream_t stream) {
  (void)in_sizes; (void)n_in; (void)out_size; (void)ws_size;

  const size_t SZ_X = (size_t)BATCH * SEQ * EMB;   // 4,194,304
  const size_t SZ_W = (size_t)EMB * EMB;           // 1,048,576

  u16* ws = (u16*)d_ws;
  u16* Xq  = ws;               u16* Xk  = Xq + SZ_X;  u16* Xv  = Xk + SZ_X;
  u16* Wqb = Xv + SZ_X;        u16* Wkb = Wqb + SZ_W; u16* Wvb = Wkb + SZ_W;
  u16* Wob = Wvb + SZ_W;
  u16* Qp  = Wob + SZ_W;       u16* Kp  = Qp + SZ_X;  u16* Vp  = Kp + SZ_X;
  u16* AO  = Vp + SZ_X;        // total 64 MB of workspace

  const int T = 256;
  // Convert activations + weights to bf16
  k_cvt_bf16<<<(int)(SZ_X / T), T, 0, stream>>>((const float*)d_in[0], Xq, (int)SZ_X);
  k_cvt_bf16<<<(int)(SZ_X / T), T, 0, stream>>>((const float*)d_in[1], Xk, (int)SZ_X);
  k_cvt_bf16<<<(int)(SZ_X / T), T, 0, stream>>>((const float*)d_in[2], Xv, (int)SZ_X);
  k_cvt_bf16<<<(int)(SZ_W / T), T, 0, stream>>>((const float*)d_in[5], Wqb, (int)SZ_W);
  k_cvt_bf16<<<(int)(SZ_W / T), T, 0, stream>>>((const float*)d_in[7], Wkb, (int)SZ_W);
  k_cvt_bf16<<<(int)(SZ_W / T), T, 0, stream>>>((const float*)d_in[9], Wvb, (int)SZ_W);
  k_cvt_bf16<<<(int)(SZ_W / T), T, 0, stream>>>((const float*)d_in[11], Wob, (int)SZ_W);

  // QKV projections (bf16 out)
  dim3 ggrid(EMB / 128, MROWS / 128);
  k_gemm<false><<<ggrid, T, 0, stream>>>(Xq, Wqb, (const float*)d_in[6],  Qp, MROWS, EMB, EMB);
  k_gemm<false><<<ggrid, T, 0, stream>>>(Xk, Wkb, (const float*)d_in[8],  Kp, MROWS, EMB, EMB);
  k_gemm<false><<<ggrid, T, 0, stream>>>(Xv, Wvb, (const float*)d_in[10], Vp, MROWS, EMB, EMB);

  // Flash attention
  dim3 fgrid(SEQ / 128, NHEAD, BATCH);
  k_flash<<<fgrid, T, 0, stream>>>(Qp, Kp, Vp, (const unsigned char*)d_in[4], AO);

  // Output projection (f32 out to d_out)
  k_gemm<true><<<ggrid, T, 0, stream>>>(AO, Wob, (const float*)d_in[12], (float*)d_out,
                                        MROWS, EMB, EMB);
}
